// ScalableMappoLagrangian_39367670235462
// MI455X (gfx1250) — matrix operand order repaired and checked by reading, not compiled, not run on hardware
//
#include <hip/hip_runtime.h>
#include <math.h>

// ---------------------------------------------------------------------------
// ScalableMappoLagrangian on MI455X (gfx1250): all dense layers via
// v_wmma_f32_16x16x32_f16 (wave32 WMMA), f16 intermediates, fused epilogues.
// B-operand staged TRANSPOSED in LDS so fragment loads are 2x ds_load_b128.
// ---------------------------------------------------------------------------

typedef __attribute__((ext_vector_type(16))) _Float16 v16h;
typedef __attribute__((ext_vector_type(8)))  _Float16 v8h;
typedef __attribute__((ext_vector_type(4)))  _Float16 h4;
typedef __attribute__((ext_vector_type(8)))  float    v8f;
typedef __attribute__((ext_vector_type(4)))  float    f4;

#define TPB   256     // 8 waves of 32
#define ROWS  128     // rows per workgroup tile
#define NTC   64      // cols per workgroup tile (4 WMMA col tiles)
#define KCH   64      // K chunk staged in LDS
#define LDSP  72      // padded LDS stride (halves): 144B -> conflict-free

#define ACT_NONE   0
#define ACT_RELU   1
#define ACT_TANH   2
#define ACT_RELUTH 3

// ---- WMMA fragment loads from LDS (contiguous k = 16*half + s, same order in A and B) ----------
// A (16x32 f16, row-major [row][k]): lanes 0-15 row=lane K {0..15};
// lanes 16-31 same row, K {16..31}. Two contiguous 8-half runs.
__device__ __forceinline__ v16h load_a_frag(const _Float16* lX, int rowBase,
                                            int ks, int lane) {
  const int half = lane >> 4, r = lane & 15;
  const _Float16* p = lX + (rowBase + r) * LDSP + ks + half * 16;
  const v8h lo = *(const v8h*)(p);        // K +0..7   -> ds_load_b128
  const v8h hi = *(const v8h*)(p + 8);    // K +8..15  -> ds_load_b128
  return __builtin_shufflevector(lo, hi, 0, 1, 2, 3, 4, 5, 6, 7,
                                 8, 9, 10, 11, 12, 13, 14, 15);
}

// B (32x16 f16) from TRANSPOSED LDS tile lWt[col][k]: lane<16 col=lane
// K=ks..ks+15, lane>=16 col=lane-16 K=ks+16..ks+31. 16 contiguous halves.
__device__ __forceinline__ v16h load_b_fragT(const _Float16* lWt, int colBase,
                                             int ks, int lane) {
  const int half = lane >> 4, c = lane & 15;
  return *(const v16h*)(lWt + (colBase + c) * LDSP + ks + half * 16);
}

// ---- Generic fused GEMM: OUT = act( [X0|X1] @ W + bias ) (+resid) ---------
// DT: 0 = f32 inputs, 1 = f16 inputs (applies to X0 and X1).
template <int DT>
__global__ __launch_bounds__(256) void gemm_mlp(
    const void* __restrict__ X0, int K0,
    const void* __restrict__ X1, int K1,
    const float* __restrict__ W, int Nfull,       // W is [K, Nfull] f32
    const float* __restrict__ bias,
    void* __restrict__ OUT, int ldo, int out16,
    int Nout, int act,
    const float* __restrict__ resid)              // stride ldo, may be null
{
  __shared__ _Float16 lX[ROWS * LDSP];   // activations, row-major [row][k]
  __shared__ _Float16 lWt[NTC * LDSP];   // weights, TRANSPOSED [col][k]

  const int t      = threadIdx.x;
  const int wave   = t >> 5;
  const int lane   = t & 31;
  const int rowBlk = blockIdx.x * ROWS;
  const int colOff = blockIdx.y * NTC;
  int ncols = Nout - colOff; if (ncols > NTC) ncols = NTC;
  const int nct = (ncols + 15) >> 4;
  const int K = K0 + K1;

  const v8f zero = {0.f, 0.f, 0.f, 0.f, 0.f, 0.f, 0.f, 0.f};
  v8f acc[4];
#pragma unroll
  for (int i = 0; i < 4; ++i) acc[i] = zero;

  for (int kc = 0; kc < K; kc += KCH) {
    // ---- stage activations: 4 contiguous halves per thread (b64 store) ----
    for (int i = t; i < ROWS * (KCH / 4); i += TPB) {
      const int rr = i >> 4;
      const int kq = (i & 15) * 4;
      const int k  = kc + kq;
      const size_t row = (size_t)(rowBlk + rr);
      h4 hv;
      if (k + 4 <= K0) {                     // fully inside X0
        if (DT) {
          hv = *(const h4*)((const _Float16*)X0 + row * K0 + k);
        } else {
          const f4 f = *(const f4*)((const float*)X0 + row * K0 + k);
          hv[0] = (_Float16)f[0]; hv[1] = (_Float16)f[1];
          hv[2] = (_Float16)f[2]; hv[3] = (_Float16)f[3];
        }
      } else if (k >= K0 && k + 4 <= K) {    // fully inside X1
        const int k1 = k - K0;
        if (DT) {
          hv = *(const h4*)((const _Float16*)X1 + row * K1 + k1);
        } else {
          const f4 f = *(const f4*)((const float*)X1 + row * K1 + k1);
          hv[0] = (_Float16)f[0]; hv[1] = (_Float16)f[1];
          hv[2] = (_Float16)f[2]; hv[3] = (_Float16)f[3];
        }
      } else {                               // ragged tail: scalar + zero pad
#pragma unroll
        for (int j = 0; j < 4; ++j) {
          const int kk = k + j;
          float v = 0.f;
          if (kk < K0)
            v = DT ? (float)((const _Float16*)X0)[row * K0 + kk]
                   : ((const float*)X0)[row * K0 + kk];
          else if (kk < K)
            v = DT ? (float)((const _Float16*)X1)[row * K1 + (kk - K0)]
                   : ((const float*)X1)[row * K1 + (kk - K0)];
          hv[j] = (_Float16)v;
        }
      }
      *(h4*)(lX + rr * LDSP + kq) = hv;
    }
    // ---- stage weights TRANSPOSED: lWt[c][k], 4 k per thread -------------
    for (int i = t; i < NTC * (KCH / 4); i += TPB) {
      const int c  = i >> 4;
      const int kq = (i & 15) * 4;
      h4 hv;
#pragma unroll
      for (int j = 0; j < 4; ++j) {
        const int k = kc + kq + j;
        float v = 0.f;
        if (k < K && c < ncols) v = W[(size_t)k * Nfull + colOff + c];
        hv[j] = (_Float16)v;
      }
      *(h4*)(lWt + c * LDSP + kq) = hv;
    }
    // prefetch next weight chunk into L2 while we compute
    if (kc + KCH < K && t < KCH)
      __builtin_prefetch((const char*)W +
          ((size_t)(kc + KCH + t) * Nfull + colOff) * sizeof(float), 0, 1);
    __syncthreads();

#pragma unroll
    for (int ks = 0; ks < KCH; ks += 32) {
      const v16h a = load_a_frag(lX, wave * 16, ks, lane);
      for (int ct = 0; ct < nct; ++ct) {
        const v16h b = load_b_fragT(lWt, ct * 16, ks, lane);
        acc[ct] = __builtin_amdgcn_wmma_f32_16x16x32_f16(
            false, a, false, b, (short)0, acc[ct], false, false);
      }
    }
    __syncthreads();
  }

  // epilogue: C/D layout — vgpr i holds row (half*8 + i), col = lane&15
  const int half = lane >> 4, r = lane & 15;
  for (int ct = 0; ct < nct; ++ct) {
    const int c = ct * 16 + r;
    if (c >= ncols) continue;
    const int colg = colOff + c;
    const float bv = bias ? bias[colg] : 0.f;
#pragma unroll
    for (int i = 0; i < 8; ++i) {
      const size_t row = (size_t)(rowBlk + wave * 16 + half * 8 + i);
      float v = acc[ct][i] + bv;
      if (act == ACT_RELU)        v = v > 0.f ? v : 0.f;
      else if (act == ACT_TANH)   v = tanhf(v);
      else if (act == ACT_RELUTH) { v -= 0.1f; v = v > 0.f ? v : 0.f; }
      if (resid) v += resid[row * (size_t)ldo + colg];
      const size_t o = row * (size_t)ldo + colg;
      if (out16) ((_Float16*)OUT)[o] = (_Float16)v;
      else       ((float*)OUT)[o]    = v;
    }
  }
}

// ---- Per-batch radius-graph mean aggregation: agg = adj @ msgs ------------
__global__ __launch_bounds__(256) void agg_kernel(
    const float* __restrict__ pos,      // [B,64,3]
    const _Float16* __restrict__ msgs,  // [B*64,64]
    _Float16* __restrict__ agg)         // [B*64,64]
{
  __shared__ float    lp[64 * 3];
  __shared__ _Float16 lA[64 * LDSP];    // adjacency, row-major [i][j]
  __shared__ _Float16 lMt[64 * LDSP];   // msgs TRANSPOSED [m][j]
  const int b = blockIdx.x;
  const int t = threadIdx.x;

  if (t < 192) lp[t] = pos[(size_t)b * 192 + t];
  // stage msgs transposed: column m, 4 consecutive j per thread
  for (int i = t; i < 64 * 16; i += TPB) {
    const int m  = i >> 4;
    const int jq = (i & 15) * 4;
    h4 hv;
#pragma unroll
    for (int j = 0; j < 4; ++j)
      hv[j] = msgs[(size_t)b * 4096 + (size_t)(jq + j) * 64 + m];
    *(h4*)(lMt + m * LDSP + jq) = hv;
  }
  __syncthreads();

  if (t < 64) {  // row t of the adjacency
    const float px = lp[t * 3], py = lp[t * 3 + 1], pz = lp[t * 3 + 2];
    int cnt = 0;
    for (int j = 0; j < 64; ++j) {
      const float dx = px - lp[j * 3];
      const float dy = py - lp[j * 3 + 1];
      const float dz = pz - lp[j * 3 + 2];
      const float d2 = dx * dx + dy * dy + dz * dz;
      const int m = (d2 < 100.0f) && (j != t);          // dist < RADIUS
      lA[t * LDSP + j] = (_Float16)(m ? 1.f : 0.f);
      cnt += m;
    }
    const float inv = 1.0f / (cnt > 0 ? (float)cnt : 1.0f);  // clip(sum,1)
    for (int j = 0; j < 64; ++j)
      lA[t * LDSP + j] = (_Float16)((float)lA[t * LDSP + j] * inv);
  }
  __syncthreads();

  const int wave = t >> 5, lane = t & 31;
  const v8f zero = {0.f, 0.f, 0.f, 0.f, 0.f, 0.f, 0.f, 0.f};
#pragma unroll
  for (int tt = 0; tt < 2; ++tt) {     // 16 output tiles, 2 per wave
    const int tile = wave * 2 + tt;
    const int mt = tile >> 2, nt = tile & 3;
    v8f c = zero;
#pragma unroll
    for (int ks = 0; ks < 64; ks += 32) {
      const v16h a  = load_a_frag(lA, mt * 16, ks, lane);
      const v16h bf = load_b_fragT(lMt, nt * 16, ks, lane);
      c = __builtin_amdgcn_wmma_f32_16x16x32_f16(
          false, a, false, bf, (short)0, c, false, false);
    }
    const int half = lane >> 4, r = lane & 15;
#pragma unroll
    for (int i = 0; i < 8; ++i) {
      const int row = mt * 16 + half * 8 + i, col = nt * 16 + r;
      agg[(size_t)b * 4096 + row * 64 + col] = (_Float16)c[i];
    }
  }
}

// ---- relu(LayerNorm(x)*g + b), f16 in/out, f32 stats; 32 lanes per row ----
__global__ __launch_bounds__(256) void ln_relu_kernel(
    const _Float16* __restrict__ in, _Float16* __restrict__ out,
    const float* __restrict__ g, const float* __restrict__ beta, int width)
{
  const int lane = threadIdx.x & 31;
  const size_t row = (size_t)blockIdx.x * 8 + (threadIdx.x >> 5);
  const _Float16* p = in + row * width;
  float s = 0.f, sq = 0.f;
  for (int c = lane; c < width; c += 32) {
    const float v = (float)p[c]; s += v; sq += v * v;
  }
  for (int off = 16; off; off >>= 1) {
    s  += __shfl_xor(s, off, 32);
    sq += __shfl_xor(sq, off, 32);
  }
  const float mean = s / width;
  const float var  = sq / width - mean * mean;
  const float rs   = rsqrtf(var + 1e-5f);
  _Float16* q = out + row * width;
  for (int c = lane; c < width; c += 32) {
    float v = ((float)p[c] - mean) * rs * g[c] + beta[c];
    q[c] = (_Float16)(v > 0.f ? v : 0.f);
  }
}

// ---- critic head: value[b] = c[b,:] . w + b0 (width 512) ------------------
__global__ __launch_bounds__(256) void critic_out_kernel(
    const _Float16* __restrict__ cin, const float* __restrict__ w,
    const float* __restrict__ b0, float* __restrict__ val)
{
  const int lane = threadIdx.x & 31;
  const size_t row = (size_t)blockIdx.x * 8 + (threadIdx.x >> 5);
  const _Float16* p = cin + row * 512;
  float s = 0.f;
  for (int c = lane; c < 512; c += 32) s += (float)p[c] * w[c];
  for (int off = 16; off; off >>= 1) s += __shfl_xor(s, off, 32);
  if (lane == 0) val[row] = s + b0[0];
}

// ---------------------------------------------------------------------------
static inline void LG(hipStream_t s, bool f16in,
                      const void* X0, int K0, const void* X1, int K1,
                      const float* W, int Nfull, const float* bias,
                      void* OUT, int ldo, bool out16, int Nout, int act,
                      const float* resid, int Mtot) {
  dim3 grid(Mtot / ROWS, (Nout + NTC - 1) / NTC);
  if (f16in)
    hipLaunchKernelGGL((gemm_mlp<1>), grid, dim3(TPB), 0, s,
        X0, K0, X1, K1, W, Nfull, bias, OUT, ldo, out16 ? 1 : 0, Nout, act, resid);
  else
    hipLaunchKernelGGL((gemm_mlp<0>), grid, dim3(TPB), 0, s,
        X0, K0, X1, K1, W, Nfull, bias, OUT, ldo, out16 ? 1 : 0, Nout, act, resid);
}

extern "C" void kernel_launch(void* const* d_in, const int* in_sizes, int n_in,
                              void* d_out, int out_size, void* d_ws, size_t ws_size,
                              hipStream_t stream) {
  (void)in_sizes; (void)n_in; (void)out_size; (void)ws_size;
  const int B = 4096, N = 64, BN = B * N;

  const float* states    = (const float*)d_in[0];
  const float* positions = (const float*)d_in[1];
  // params flattened depth-first in setup_inputs() dict order:
  // attn.enc.l1{w,b} attn.enc.l2{w,b} attn.aggr.l1 attn.aggr.l2
  // attn.proj.l1 attn.proj.l2 | actor.lin[0..2]{w,b} actor.ln_g[0..2]
  // actor.ln_b[0..2] actor.out{w,b} | critic likewise | constraint l1..l3
  auto P = [&](int i) { return (const float*)d_in[i]; };

  // workspace carve-up
  char* ws = (char*)d_ws;
  size_t off = 0;
  auto take = [&](size_t bytes) {
    void* p = ws + off; off += (bytes + 255) & ~(size_t)255; return p;
  };
  _Float16* h1     = (_Float16*)take((size_t)BN * 256 * 2); // 134 MB
  _Float16* h2     = (_Float16*)take((size_t)BN * 256 * 2); // 134 MB
  _Float16* msgs   = (_Float16*)take((size_t)BN * 64 * 2);  //  33 MB
  _Float16* aggb   = (_Float16*)take((size_t)BN * 64 * 2);  //  33 MB
  float*    finalb = (float*)   take((size_t)BN * 64 * 4);  //  67 MB
  float*    xcur   = (float*)   take((size_t)BN * 64 * 4);  //  67 MB
  _Float16* c1     = (_Float16*)take((size_t)B * 512 * 2);  //   4 MB
  _Float16* c2     = (_Float16*)take((size_t)B * 512 * 2);  //   4 MB

  float* actOut  = (float*)d_out;                 // [B,N,8]
  float* valOut  = actOut + (size_t)BN * 8;       // [B,1]
  float* violOut = valOut + B;                    // [B,N,4]

  // ---- 2 communication rounds ----
  for (int r = 0; r < 2; ++r) {
    const float* xin = r ? xcur : states;
    LG(stream, false, xin, 64, nullptr, 0, P(2), 256, P(3),
       h1, 256, true, 256, ACT_RELU, nullptr, BN);                 // enc.l1
    LG(stream, true, h1, 256, nullptr, 0, P(4), 64, P(5),
       msgs, 64, true, 64, ACT_NONE, nullptr, BN);                 // enc.l2
    hipLaunchKernelGGL(agg_kernel, dim3(B), dim3(TPB), 0, stream,
                       positions, msgs, aggb);                     // adj@msgs
    LG(stream, true, msgs, 64, aggb, 64, P(6), 256, P(7),
       h1, 256, true, 256, ACT_RELU, nullptr, BN);                 // aggr.l1
    LG(stream, true, h1, 256, nullptr, 0, P(8), 64, P(9),
       finalb, 64, false, 64, ACT_NONE, nullptr, BN);              // aggr.l2
    LG(stream, false, xin, 64, finalb, 64, P(10), 256, P(11),
       h1, 256, true, 256, ACT_RELU, nullptr, BN);                 // proj.l1
    LG(stream, true, h1, 256, nullptr, 0, P(12), 64, P(13),
       xcur, 64, false, 64, ACT_NONE, xin, BN);                    // proj.l2 + x
  }

  // ---- actor ----
  LG(stream, false, xcur, 64, nullptr, 0, P(14), 256, P(15),
     h1, 256, true, 256, ACT_NONE, nullptr, BN);
  hipLaunchKernelGGL(ln_relu_kernel, dim3(BN / 8), dim3(TPB), 0, stream,
                     h1, h2, P(20), P(23), 256);
  LG(stream, true, h2, 256, nullptr, 0, P(16), 256, P(17),
     h1, 256, true, 256, ACT_NONE, nullptr, BN);
  hipLaunchKernelGGL(ln_relu_kernel, dim3(BN / 8), dim3(TPB), 0, stream,
                     h1, h2, P(21), P(24), 256);
  LG(stream, true, h2, 256, nullptr, 0, P(18), 256, P(19),
     h1, 256, true, 256, ACT_NONE, nullptr, BN);
  hipLaunchKernelGGL(ln_relu_kernel, dim3(BN / 8), dim3(TPB), 0, stream,
                     h1, h2, P(22), P(25), 256);
  LG(stream, true, h2, 256, nullptr, 0, P(26), 16, P(27),
     actOut, 8, false, 8, ACT_TANH, nullptr, BN);   // tanh(actor_out[:,:8])

  // ---- centralized critic on raw states viewed as [B, N*D] ----
  LG(stream, false, states, 4096, nullptr, 0, P(28), 512, P(29),
     c1, 512, true, 512, ACT_NONE, nullptr, B);
  hipLaunchKernelGGL(ln_relu_kernel, dim3(B / 8), dim3(TPB), 0, stream,
                     c1, c2, P(34), P(37), 512);
  LG(stream, true, c2, 512, nullptr, 0, P(30), 512, P(31),
     c1, 512, true, 512, ACT_NONE, nullptr, B);
  hipLaunchKernelGGL(ln_relu_kernel, dim3(B / 8), dim3(TPB), 0, stream,
                     c1, c2, P(35), P(38), 512);
  LG(stream, true, c2, 512, nullptr, 0, P(32), 512, P(33),
     c1, 512, true, 512, ACT_NONE, nullptr, B);
  hipLaunchKernelGGL(ln_relu_kernel, dim3(B / 8), dim3(TPB), 0, stream,
                     c1, c2, P(36), P(39), 512);
  hipLaunchKernelGGL(critic_out_kernel, dim3(B / 8), dim3(TPB), 0, stream,
                     c2, P(40), P(41), valOut);

  // ---- constraint net on concat(states, actions) ----
  LG(stream, false, states, 64, actOut, 8, P(42), 256, P(43),
     h1, 256, true, 256, ACT_RELU, nullptr, BN);
  LG(stream, true, h1, 256, nullptr, 0, P(44), 256, P(45),
     h2, 256, true, 256, ACT_RELU, nullptr, BN);
  LG(stream, true, h2, 256, nullptr, 0, P(46), 4, P(47),
     violOut, 4, false, 4, ACT_RELUTH, nullptr, BN); // relu(v - 0.1)
}
